// MultilevelSelfAttentionBlockWithRoPE_55929064129201
// MI455X (gfx1250) — matrix operand order repaired and checked by reading, not compiled, not run on hardware
//
#include <hip/hip_runtime.h>
#include <hip/hip_bf16.h>

// ---------------------------------------------------------------------------
// MultilevelSelfAttentionBlockWithRoPE for MI455X (gfx1250, wave32, WMMA)
// ---------------------------------------------------------------------------
#define N_TOKENS   7168
#define EMBED_DIM  256
#define N_HEADS    8
#define HEAD_DIM   32
#define HALF_DIM   16
#define BATCH      4
#define MAX_LEN    2048
#define QKV_DIM    768   // 3 * EMBED_DIM
#define P_LD       40    // padded LDS row stride (halves): 80B, 16B-aligned

typedef __attribute__((ext_vector_type(16))) _Float16 v16h;
typedef __attribute__((ext_vector_type(8)))  _Float16 v8h;
typedef __attribute__((ext_vector_type(8)))  float    v8f;

// ---- WMMA fragment loaders (layouts per CDNA5 ISA 7.12.2, wave32) ----------
// A fragment 16x32 f16: lane L holds row m=L&15; hi=L>>4 selects K-split.
//   VGPR0..3 -> K = (hi?8:0)  + 0..7   (contiguous)
//   VGPR4..7 -> K = (hi?24:16)+ 0..7   (contiguous)
__device__ __forceinline__ v16h load_a_frag(const _Float16* src, int ld,
                                            int row0, int k0) {
  const int lane = threadIdx.x & 31;
  const int m  = lane & 15;
  const int hi = lane >> 4;
  const _Float16* p = src + (size_t)(row0 + m) * ld + k0 + hi * 8;
  v8h lo = *(const v8h*)p;
  v8h hh = *(const v8h*)(p + 16);
  v16h r;
#pragma unroll
  for (int i = 0; i < 8; ++i) { r[i] = lo[i]; r[8 + i] = hh[i]; }
  return r;
}

// B fragment 32x16 f16 loaded from a row-major matrix W where W[n][k] == B[k][n]
// (i.e. B-transposed storage). Lane L holds col n=L&15; hi=L>>4 selects
// K = (hi?8:0) + 0..7 and (hi?24:16) + 0..7 (same order as A) -> two 16-byte loads.
__device__ __forceinline__ v16h load_bT_frag(const _Float16* w, int ld,
                                             int n0, int k0) {
  const int lane = threadIdx.x & 31;
  const int n  = lane & 15;
  const int hi = lane >> 4;
  const _Float16* p = w + (size_t)(n0 + n) * ld + k0 + hi * 8;
  v8h lo = *(const v8h*)p;
  v8h hh = *(const v8h*)(p + 16);
  v16h r;
#pragma unroll
  for (int i = 0; i < 8; ++i) { r[i] = lo[i]; r[8 + i] = hh[i]; }
  return r;
}

__device__ __forceinline__ v8f wmma_f16(v16h a, v16h b, v8f c) {
  return __builtin_amdgcn_wmma_f32_16x16x32_f16(false, a, false, b,
                                                (short)0, c, false, false);
}

// ---------------------------------------------------------------------------
// 1) LayerNorm: x fp32 -> xn f16
// ---------------------------------------------------------------------------
__global__ void ln_kernel(const float* __restrict__ x,
                          const float* __restrict__ w,
                          const float* __restrict__ b,
                          _Float16* __restrict__ xnh) {
  const int n = blockIdx.x;
  const int c = threadIdx.x;            // 256 threads = 8 waves
  float v = x[(size_t)n * EMBED_DIM + c];
  float s = v, s2 = v * v;
#pragma unroll
  for (int off = 16; off > 0; off >>= 1) {
    s  += __shfl_down(s,  off, 32);
    s2 += __shfl_down(s2, off, 32);
  }
  __shared__ float red[2][8];
  const int wid = threadIdx.x >> 5, lid = threadIdx.x & 31;
  if (lid == 0) { red[0][wid] = s; red[1][wid] = s2; }
  __syncthreads();
  if (wid == 0) {
    float a  = (lid < 8) ? red[0][lid] : 0.f;
    float a2 = (lid < 8) ? red[1][lid] : 0.f;
#pragma unroll
    for (int off = 4; off > 0; off >>= 1) {
      a += __shfl_down(a, off, 32); a2 += __shfl_down(a2, off, 32);
    }
    if (lid == 0) { red[0][0] = a; red[1][0] = a2; }
  }
  __syncthreads();
  const float mean = red[0][0] * (1.f / EMBED_DIM);
  const float var  = red[1][0] * (1.f / EMBED_DIM) - mean * mean;
  const float r    = rsqrtf(var + 1e-5f);
  xnh[(size_t)n * EMBED_DIM + c] = (_Float16)((v - mean) * r * w[c] + b[c]);
}

// ---------------------------------------------------------------------------
// 2) Weight conversion fp32 -> f16 (qkv_w 768x256, out_w 256x256)
// ---------------------------------------------------------------------------
__global__ void cvt_w_kernel(const float* __restrict__ qw,
                             const float* __restrict__ ow,
                             _Float16* __restrict__ qwh,
                             _Float16* __restrict__ owh) {
  const int i = blockIdx.x * 256 + threadIdx.x;
  if (i < QKV_DIM * EMBED_DIM)   qwh[i] = (_Float16)qw[i];
  if (i < EMBED_DIM * EMBED_DIM) owh[i] = (_Float16)ow[i];
}

// ---------------------------------------------------------------------------
// 3) QKV GEMM: qkv[N][768] = xn[N][256] @ qkv_w^T   (W stored row-major 768x256)
//    4 waves/block; each wave owns one 16x16 tile; 8 WMMA along K.
// ---------------------------------------------------------------------------
__global__ void gemm_qkv_kernel(const _Float16* __restrict__ A,
                                const _Float16* __restrict__ W,
                                float* __restrict__ C) {
  const int wave = threadIdx.x >> 5;
  const int mt = blockIdx.x;                  // 0..447
  const int nt = blockIdx.y * 4 + wave;       // 0..47
  const int row0 = mt * 16, col0 = nt * 16;
  v8f acc = {};
#pragma unroll
  for (int k = 0; k < EMBED_DIM; k += 32) {
    v16h a = load_a_frag(A, EMBED_DIM, row0, k);
    v16h b = load_bT_frag(W, EMBED_DIM, col0, k);
    acc = wmma_f16(a, b, acc);
  }
  const int lane = threadIdx.x & 31;
  const int cn = lane & 15, hi = lane >> 4;
#pragma unroll
  for (int v = 0; v < 8; ++v)
    C[(size_t)(row0 + v + hi * 8) * QKV_DIM + col0 + cn] = acc[v];
}

// ---------------------------------------------------------------------------
// 4) RoPE + scatter to padded head-major layouts.
//    qpad/kpad: [b][h][p][d] f16 (q pre-scaled by 1/sqrt(32))
//    vT:        [b][h][d][p] f16 (transposed: A-fragment layout for O^T = V^T P)
// ---------------------------------------------------------------------------
__global__ void rope_scatter_kernel(const float* __restrict__ qkv,
                                    const float* __restrict__ spatial,
                                    const int*   __restrict__ lvl,
                                    const int*   __restrict__ shapes,
                                    const int*   __restrict__ boff,
                                    const float* __restrict__ freqs,
                                    _Float16* __restrict__ qpad,
                                    _Float16* __restrict__ kpad,
                                    _Float16* __restrict__ vT) {
  const int n = blockIdx.x;
  const int t = threadIdx.x;               // 128 threads: (h,f) pairs
  int b = 0;
#pragma unroll
  for (int i = 1; i < BATCH; ++i) if (n >= boff[i]) b = i;
  const int p  = n - boff[b];
  const int li = lvl[n];
  const float sh0 = (float)shapes[li * 2 + 0];
  const float sh1 = (float)shapes[li * 2 + 1];
  int mx0 = shapes[0], mx1 = shapes[1];
#pragma unroll
  for (int i = 1; i < 4; ++i) {
    int a0 = shapes[i * 2], a1 = shapes[i * 2 + 1];
    mx0 = (a0 > mx0) ? a0 : mx0;
    mx1 = (a1 > mx1) ? a1 : mx1;
  }
  const float p0 = spatial[(size_t)n * 2 + 0] * ((float)mx0 / sh0);
  const float p1 = spatial[(size_t)n * 2 + 1] * ((float)mx1 / sh1);
  const float p2 = (float)li;

  const int h = t >> 4, f = t & 15;
  const float th = p0 * freqs[0 * 128 + h * 16 + f]
                 + p1 * freqs[1 * 128 + h * 16 + f]
                 + p2 * freqs[2 * 128 + h * 16 + f];
  const float c = cosf(th), s = sinf(th);

  const float* q = qkv + (size_t)n * QKV_DIM;
  const float* k = q + EMBED_DIM;
  const float* v = k + EMBED_DIM;
  const float q0 = q[h * 32 + 2 * f], q1 = q[h * 32 + 2 * f + 1];
  const float k0 = k[h * 32 + 2 * f], k1 = k[h * 32 + 2 * f + 1];
  const float qs = 0.17677669529663687f;   // 1/sqrt(HEAD_DIM)
  const size_t base = ((size_t)(b * N_HEADS + h) * MAX_LEN + p) * HEAD_DIM;
  qpad[base + 2 * f]     = (_Float16)((q0 * c - q1 * s) * qs);
  qpad[base + 2 * f + 1] = (_Float16)((q0 * s + q1 * c) * qs);
  kpad[base + 2 * f]     = (_Float16)(k0 * c - k1 * s);
  kpad[base + 2 * f + 1] = (_Float16)(k0 * s + k1 * c);
#pragma unroll
  for (int e = 0; e < 2; ++e) {
    const int col = t * 2 + e;             // 0..255
    const int hh = col >> 5, d = col & 31;
    vT[((size_t)(b * N_HEADS + hh) * HEAD_DIM + d) * MAX_LEN + p] =
        (_Float16)v[col];
  }
}

// ---------------------------------------------------------------------------
// 5) Flash attention, transposed-score formulation (softmax keys in-lane):
//    S^T(16k x 16q) = K_tile(A) x Q^T(B)   -> keys indexed by (v,hi) in-lane
//    O^T(32d x 16q) = V^T(A)    x P(B)     -> per-lane stats, b128 epilogue
//    1 wave per 16-query tile, 32 keys per chunk, 4 WMMA per chunk.
// ---------------------------------------------------------------------------
__global__ void attn_kernel(const _Float16* __restrict__ qpad,
                            const _Float16* __restrict__ kpad,
                            const _Float16* __restrict__ vT,
                            const int* __restrict__ boff,
                            _Float16* __restrict__ attnh) {
  __shared__ __align__(32) _Float16 Pl[4][16 * P_LD];   // 1.25KB per wave
  const int wave = threadIdx.x >> 5;
  const int lane = threadIdx.x & 31;
  const int tile = blockIdx.x * 4 + wave;   // 0..4095
  const int b  = tile >> 10;                // 1024 tiles per batch
  const int h  = (tile >> 7) & 7;
  const int qt = tile & 127;
  const int len = boff[b + 1] - boff[b];
  const int q0 = qt * 16;
  if (q0 >= len) return;                    // wave-uniform exit

  const _Float16* qb = qpad + ((size_t)(b * N_HEADS + h) * MAX_LEN + q0) * HEAD_DIM;
  const _Float16* kb = kpad + ((size_t)(b * N_HEADS + h) * MAX_LEN) * HEAD_DIM;
  const _Float16* vb = vT   + ((size_t)(b * N_HEADS + h) * HEAD_DIM) * MAX_LEN;

  // Q as the B operand: lane = q column, contiguous d.
  const v16h qf = load_bT_frag(qb, HEAD_DIM, 0, 0);
  const int cn = lane & 15, hi = lane >> 4;
  _Float16* pl = &Pl[wave][0];

  float mrun = -3.0e38f, lrun = 0.f;        // per-lane (per q column) stats
  v8f o0 = {}, o1 = {};                     // O^T: rows d = hi*8+v (+16)

  for (int kc = 0; kc < len; kc += 32) {
    if (kc + 32 < len)
      __builtin_prefetch(kb + (size_t)(kc + 32) * HEAD_DIM, 0, 1);
    // K tiles as A operand: lane = key row, contiguous d.
    v16h ka0 = load_a_frag(kb + (size_t)kc * HEAD_DIM,        HEAD_DIM, 0, 0);
    v16h ka1 = load_a_frag(kb + (size_t)(kc + 16) * HEAD_DIM, HEAD_DIM, 0, 0);
    v8f z = {};
    v8f s0 = wmma_f16(ka0, qf, z);          // S^T keys kc+0..15
    v8f s1 = wmma_f16(ka1, qf, z);          // S^T keys kc+16..31
    if (kc + 32 > len) {                    // partial chunk: mask padded keys
#pragma unroll
      for (int v = 0; v < 8; ++v) {
        if (kc + hi * 8 + v      >= len) s0[v] = -1e9f;
        if (kc + 16 + hi * 8 + v >= len) s1[v] = -1e9f;
      }
    }
    // chunk max: 15 in-lane fmax + one cross-half shuffle
    float mv = fmaxf(s0[0], s1[0]);
#pragma unroll
    for (int v = 1; v < 8; ++v) mv = fmaxf(mv, fmaxf(s0[v], s1[v]));
    mv = fmaxf(mv, __shfl_xor(mv, 16, 32));
    const float mnew  = fmaxf(mrun, mv);
    const float alpha = __expf(mrun - mnew);
    mrun = mnew;
    // exponentiate, in-lane partial sum, stage P^T -> LDS P[q][key]
    float ls = 0.f;
#pragma unroll
    for (int v = 0; v < 8; ++v) {
      const float p0e = __expf(s0[v] - mnew);
      const float p1e = __expf(s1[v] - mnew);
      ls += p0e + p1e;
      pl[cn * P_LD + hi * 8 + v]      = (_Float16)p0e;
      pl[cn * P_LD + 16 + hi * 8 + v] = (_Float16)p1e;
      o0[v] *= alpha;
      o1[v] *= alpha;
    }
    ls += __shfl_xor(ls, 16, 32);
    lrun = lrun * alpha + ls;
    // O^T += V^T(A) x P(B):  P read back with lane = q, contiguous keys.
    const v16h pf  = load_bT_frag(pl, P_LD, 0, 0);
    const v16h va0 = load_a_frag(vb, MAX_LEN, 0,  kc);
    const v16h va1 = load_a_frag(vb, MAX_LEN, 16, kc);
    o0 = wmma_f16(va0, pf, o0);
    o1 = wmma_f16(va1, pf, o1);
  }
  // epilogue: per-lane normalize, two contiguous v8h (b128) stores per lane
  if (q0 + cn < len) {
    const float inv = 1.0f / lrun;
    v8h r0, r1;
#pragma unroll
    for (int v = 0; v < 8; ++v) {
      r0[v] = (_Float16)(o0[v] * inv);
      r1[v] = (_Float16)(o1[v] * inv);
    }
    const size_t idx = (size_t)(boff[b] + q0 + cn) * EMBED_DIM + h * HEAD_DIM;
    *(v8h*)(attnh + idx + hi * 8)      = r0;   // d = hi*8 + 0..7
    *(v8h*)(attnh + idx + 16 + hi * 8) = r1;   // d = 16 + hi*8 + 0..7
  }
}

// ---------------------------------------------------------------------------
// 6) Output GEMM + residual: out[N][256] = attnh @ out_w^T + x
// ---------------------------------------------------------------------------
__global__ void gemm_out_kernel(const _Float16* __restrict__ A,
                                const _Float16* __restrict__ W,
                                const float* __restrict__ residual,
                                float* __restrict__ out) {
  const int wave = threadIdx.x >> 5;
  const int mt = blockIdx.x;                 // 0..447
  const int nt = blockIdx.y * 4 + wave;      // 0..15
  const int row0 = mt * 16, col0 = nt * 16;
  v8f acc = {};
#pragma unroll
  for (int k = 0; k < EMBED_DIM; k += 32) {
    v16h a = load_a_frag(A, EMBED_DIM, row0, k);
    v16h b = load_bT_frag(W, EMBED_DIM, col0, k);
    acc = wmma_f16(a, b, acc);
  }
  const int lane = threadIdx.x & 31;
  const int cn = lane & 15, hi = lane >> 4;
#pragma unroll
  for (int v = 0; v < 8; ++v) {
    const size_t idx = (size_t)(row0 + v + hi * 8) * EMBED_DIM + col0 + cn;
    out[idx] = acc[v] + residual[idx];
  }
}

// ---------------------------------------------------------------------------
extern "C" void kernel_launch(void* const* d_in, const int* in_sizes, int n_in,
                              void* d_out, int out_size, void* d_ws, size_t ws_size,
                              hipStream_t stream) {
  const float* x       = (const float*)d_in[0];
  const float* spatial = (const float*)d_in[1];
  const int*   lvl     = (const int*)  d_in[2];
  const int*   shapes  = (const int*)  d_in[3];
  const int*   boff    = (const int*)  d_in[4];
  const float* lnw     = (const float*)d_in[5];
  const float* lnb     = (const float*)d_in[6];
  const float* qw      = (const float*)d_in[7];
  const float* ow      = (const float*)d_in[8];
  const float* freqs   = (const float*)d_in[9];
  float* out = (float*)d_out;

  char* ws = (char*)d_ws;
  size_t off = 0;
  auto walloc = [&](size_t bytes) -> void* {
    void* p = ws + off;
    off += (bytes + 255) & ~(size_t)255;
    return p;
  };
  _Float16* xnh   = (_Float16*)walloc((size_t)N_TOKENS * EMBED_DIM * 2);
  _Float16* qwh   = (_Float16*)walloc((size_t)QKV_DIM * EMBED_DIM * 2);
  _Float16* owh   = (_Float16*)walloc((size_t)EMBED_DIM * EMBED_DIM * 2);
  float*    qkv   = (float*)   walloc((size_t)N_TOKENS * QKV_DIM * 4);
  _Float16* qpad  = (_Float16*)walloc((size_t)BATCH * N_HEADS * MAX_LEN * HEAD_DIM * 2);
  _Float16* kpad  = (_Float16*)walloc((size_t)BATCH * N_HEADS * MAX_LEN * HEAD_DIM * 2);
  _Float16* vT    = (_Float16*)walloc((size_t)BATCH * N_HEADS * MAX_LEN * HEAD_DIM * 2);
  _Float16* attnh = (_Float16*)walloc((size_t)N_TOKENS * EMBED_DIM * 2);

  ln_kernel<<<N_TOKENS, EMBED_DIM, 0, stream>>>(x, lnw, lnb, xnh);
  cvt_w_kernel<<<(QKV_DIM * EMBED_DIM) / 256, 256, 0, stream>>>(qw, ow, qwh, owh);
  gemm_qkv_kernel<<<dim3(N_TOKENS / 16, QKV_DIM / 16 / 4), 128, 0, stream>>>(xnh, qwh, qkv);
  rope_scatter_kernel<<<N_TOKENS, 128, 0, stream>>>(qkv, spatial, lvl, shapes,
                                                    boff, freqs, qpad, kpad, vT);
  attn_kernel<<<(BATCH * N_HEADS * (MAX_LEN / 16)) / 4, 128, 0, stream>>>(
      qpad, kpad, vT, boff, attnh);
  gemm_out_kernel<<<dim3(N_TOKENS / 16, EMBED_DIM / 16 / 4), 128, 0, stream>>>(
      attnh, owh, x, out);
}
